// DNC_2783138808151
// MI455X (gfx1250) — compile-verified, matrix operand order checked by reading, not run on hardware
//
#include <hip/hip_runtime.h>
#include <hip/hip_bf16.h>
#include <math.h>

typedef __attribute__((ext_vector_type(16))) _Float16 v16h;
typedef __attribute__((ext_vector_type(8)))  _Float16 v8h;
typedef __attribute__((ext_vector_type(8)))  float    v8f;

#define Bdim 8
#define Tdim 1024
#define Ddim 256
#define Hdim 512
#define Mdim 1024
#define Rdim 4

__device__ __forceinline__ float sigf(float x) { return 1.0f / (1.0f + __expf(-x)); }

// Load one 16-element WMMA fragment (per-lane) from a K-contiguous row
// segment: elements 0..7 at p[0..7], elements 8..15 at p[16..23].
// p must be 16B aligned -> two b128 loads.
__device__ __forceinline__ v16h frag16(const _Float16* p) {
  v8h lo = *(const v8h*)(p);
  v8h hi = *(const v8h*)(p + 16);
  return __builtin_shufflevector(lo, hi, 0, 1, 2, 3, 4, 5, 6, 7,
                                 8, 9, 10, 11, 12, 13, 14, 15);
}

// ---------------------------------------------------------------------------
// Tiled WMMA GEMM:  C[M,N] = act(A[M,K] @ W[K,N] + bias[N])
// fp32 global -> f16 LDS staging, fp32 accumulate (v_wmma_f32_16x16x32_f16).
// Block = 256 threads = 8 waves (2x4); block tile 32x128; wave tile 16x32
// (2 WMMA sharing one A fragment); K-step 32.  B tile staged TRANSPOSED so
// both A and B fragments are contiguous 2x16B per lane (ds_load_b128).
// M%32==0, N%128==0, K%32==0 assumed.  act: 0=none, 1=sigmoid.
// ---------------------------------------------------------------------------
__global__ void gemm_wmma_f16(const float* __restrict__ A,
                              const float* __restrict__ W,
                              const float* __restrict__ bias,
                              float* __restrict__ C,
                              int Mrows, int Ncols, int Kdim, int act) {
  __shared__ _Float16 As[32][40];    // [m][k]   row pitch 80B (16B aligned)
  __shared__ _Float16 BsT[128][40];  // [n][k]   transposed

  const int tid  = threadIdx.x;
  const int lane = tid & 31;
  const int wave = tid >> 5;
  const int wm   = wave >> 2;          // 0..1
  const int wn   = wave & 3;           // 0..3
  const int row0 = blockIdx.x * 32;
  const int col0 = blockIdx.y * 128;

  const int fm = lane & 15;            // fragment M (A) / N (B)
  const int kh = lane >> 4;            // K-half selector

  v8f acc[2] = {};

  for (int k0 = 0; k0 < Kdim; k0 += 32) {
    if (k0 + 32 < Kdim) {              // emits global_prefetch_b8
      __builtin_prefetch(&A[(size_t)(row0 + (tid >> 3)) * Kdim + k0 + 32], 0, 0);
      __builtin_prefetch(&W[(size_t)(k0 + 32 + (tid >> 3)) * Ncols + col0], 0, 0);
    }
    // stage A tile 32x32 (4/thread), coalesced global, row-major LDS
#pragma unroll
    for (int i = 0; i < 4; ++i) {
      int e = tid + i * 256;
      int r = e >> 5, c = e & 31;
      As[r][c] = (_Float16)A[(size_t)(row0 + r) * Kdim + (k0 + c)];
    }
    // stage B tile 32x128 (16/thread), coalesced global, TRANSPOSED LDS
#pragma unroll
    for (int i = 0; i < 16; ++i) {
      int e = tid + i * 256;
      int c = e & 127, r = e >> 7;
      BsT[c][r] = (_Float16)W[(size_t)(k0 + r) * Ncols + (col0 + c)];
    }
    __syncthreads();

    v16h af = frag16(&As[wm * 16 + fm][kh * 8]);
#pragma unroll
    for (int h = 0; h < 2; ++h) {
      v16h bf = frag16(&BsT[wn * 32 + h * 16 + fm][kh * 8]);
      acc[h] = __builtin_amdgcn_wmma_f32_16x16x32_f16(false, af, false, bf,
                                                      (short)0, acc[h],
                                                      false, false);
    }
    __syncthreads();
  }

  // writeback: C/D layout — lane n = lane&15, VGPR i -> m = i + 8*(lane>>4)
  const int on = lane & 15;
#pragma unroll
  for (int h = 0; h < 2; ++h) {
    int col = col0 + wn * 32 + h * 16 + on;
    float bv = bias ? bias[col] : 0.0f;
#pragma unroll
    for (int i = 0; i < 8; ++i) {
      int om = i + (kh << 3);
      float v = acc[h][i] + bv;
      if (act == 1) v = sigf(v);
      C[(size_t)(row0 + wm * 16 + om) * Ncols + col] = v;
    }
  }
}

// ---------------------------------------------------------------------------
// Wh (H x 4H fp32, row-major) -> WhT (4H x H f16, transposed) so LSTM
// B-fragments are contiguous global_load_b128 per lane.
// ---------------------------------------------------------------------------
__global__ void cvt_transpose_f16(const float* __restrict__ in,   // (K=512, N=2048)
                                  _Float16* __restrict__ out) {   // (N=2048, K=512)
  int i = blockIdx.x * 256 + threadIdx.x;      // over 2048*512
  if (i < Hdim * 4 * Hdim) {
    int n = i >> 9;          // 0..2047
    int k = i & 511;         // 0..511
    out[(size_t)n * Hdim + k] = (_Float16)in[(size_t)k * (4 * Hdim) + n];
  }
}

// ---------------------------------------------------------------------------
// Persistent single-block LSTM recurrence.
// Per step: z = XZ[:,t,:] + h @ Wh via WMMA.  h (f16, 16-row padded) in LDS;
// c in registers.  Each wave: 4 gate-chunks x (16 k-steps x 4 WMMA sharing
// one A fragment) = 256 WMMA/step, 2048 WMMA/step total across 8 waves.
// ---------------------------------------------------------------------------
__global__ void lstm_seq(const float* __restrict__ XZ,       // (B*T, 4H)
                         const _Float16* __restrict__ WhT,   // (2048, 512) f16
                         float* __restrict__ Hout) {         // (B*T, 512)
  __shared__ _Float16 hbuf[16][Hdim];        // rows 8..15 stay zero (pad)
  __shared__ _Float16 zbuf[Bdim][4 * Hdim];  // 8 x 2048

  const int tid  = threadIdx.x;
  const int lane = tid & 31;
  const int wave = tid >> 5;
  const int fm   = lane & 15;
  const int kh   = lane >> 4;

  for (int i = tid; i < 16 * Hdim; i += 256)
    ((_Float16*)hbuf)[i] = (_Float16)0.0f;

  float c[16];
#pragma unroll
  for (int i = 0; i < 16; ++i) c[i] = 0.0f;
  __syncthreads();

  for (int t = 0; t < Tdim; ++t) {
    // --- z = h @ Wh ---
    for (int q = 0; q < 4; ++q) {              // gate chunk
      const int nbase = q * Hdim + wave * 64;
      v8f acc[4] = {};
      for (int k0 = 0; k0 < Hdim; k0 += 32) {
        v16h af = frag16(&hbuf[fm][k0 + kh * 8]);
#pragma unroll
        for (int j = 0; j < 4; ++j) {
          v16h bf = frag16(WhT + (size_t)(nbase + j * 16 + fm) * Hdim
                               + k0 + kh * 8);
          acc[j] = __builtin_amdgcn_wmma_f32_16x16x32_f16(false, af, false, bf,
                                                          (short)0, acc[j],
                                                          false, false);
        }
      }
#pragma unroll
      for (int j = 0; j < 4; ++j) {
        int n0 = nbase + j * 16;
#pragma unroll
        for (int i = 0; i < 8; ++i) {
          int row = i + (kh << 3);
          if (row < Bdim) {
            float xz = XZ[((size_t)row * Tdim + t) * (4 * Hdim) + (n0 + fm)];
            zbuf[row][n0 + fm] = (_Float16)(acc[j][i] + xz);
          }
        }
      }
    }
    __syncthreads();

    // --- gates: 8*512 = 4096 states, 16 per thread ---
#pragma unroll
    for (int i = 0; i < 16; ++i) {
      int idx = tid * 16 + i;
      int b   = idx >> 9;
      int n   = idx & 511;
      float zi = (float)zbuf[b][n];
      float zf = (float)zbuf[b][n + Hdim];
      float zg = (float)zbuf[b][n + 2 * Hdim];
      float zo = (float)zbuf[b][n + 3 * Hdim];
      float cn = sigf(zf) * c[i] + sigf(zi) * tanhf(zg);
      c[i] = cn;
      float h = sigf(zo) * tanhf(cn);
      hbuf[b][n] = (_Float16)h;
      Hout[((size_t)b * Tdim + t) * Hdim + n] = h;
    }
    __syncthreads();
  }
}

// ---------------------------------------------------------------------------
// block reductions
// ---------------------------------------------------------------------------
__device__ __forceinline__ float blk_max(float v, float* red) {
  int tid = threadIdx.x;
  red[tid] = v; __syncthreads();
  for (int s = 128; s > 0; s >>= 1) {
    if (tid < s) red[tid] = fmaxf(red[tid], red[tid + s]);
    __syncthreads();
  }
  float r = red[0]; __syncthreads();
  return r;
}
__device__ __forceinline__ float blk_sum(float v, float* red) {
  int tid = threadIdx.x;
  red[tid] = v; __syncthreads();
  for (int s = 128; s > 0; s >>= 1) {
    if (tid < s) red[tid] += red[tid + s];
    __syncthreads();
  }
  float r = red[0]; __syncthreads();
  return r;
}

// per-row 1/||memory[t]|| and column sums of memory
__global__ void mem_invnorm(const float* __restrict__ mem, float* __restrict__ invn) {
  __shared__ float red[256];
  int t = blockIdx.x;
  float s = 0.0f;
  for (int j = threadIdx.x; j < Mdim; j += 256) {
    float v = mem[(size_t)t * Mdim + j];
    s += v * v;
  }
  float tot = blk_sum(s, red);
  if (threadIdx.x == 0) invn[t] = rsqrtf(fmaxf(tot, 1e-12f));
}
__global__ void mem_colsum(const float* __restrict__ mem, float* __restrict__ cs) {
  int j = blockIdx.x * 256 + threadIdx.x;
  if (j < Mdim) {
    float s = 0.0f;
    for (int i = 0; i < Mdim; ++i) s += mem[(size_t)i * Mdim + j];
    cs[j] = s;
  }
}

// fused: row softmax -> l2norm -> (negative) dot with l2norm(memory[t]).
// one block per row; t = row % T.
__global__ void softmax_l2_dot(const float* __restrict__ logits,
                               const float* __restrict__ mem,
                               const float* __restrict__ invn,
                               float* __restrict__ sim) {
  __shared__ float red[256];
  int row = blockIdx.x;
  int t   = row & (Tdim - 1);
  const float* x = logits + (size_t)row * Mdim;

  float mx = -1e30f;
  for (int j = threadIdx.x; j < Mdim; j += 256) mx = fmaxf(mx, x[j]);
  float rowmax = blk_max(mx, red);

  float se = 0.0f;
  for (int j = threadIdx.x; j < Mdim; j += 256) se += __expf(x[j] - rowmax);
  float denom = blk_sum(se, red);
  float inv_d = 1.0f / denom;

  float ss = 0.0f, sd = 0.0f;
  float inv_n = invn[t];
  for (int j = threadIdx.x; j < Mdim; j += 256) {
    float s = __expf(x[j] - rowmax) * inv_d;
    ss += s * s;
    sd += s * mem[(size_t)t * Mdim + j] * inv_n;
  }
  float sst = blk_sum(ss, red);
  float sdt = blk_sum(sd, red);
  if (threadIdx.x == 0) sim[row] = -sdt * rsqrtf(fmaxf(sst, 1e-12f));
}

// softmax over T (length 1024) per block; optional per-element scale[t]
__global__ void softmax_T(const float* __restrict__ sim,
                          const float* __restrict__ scale,
                          float* __restrict__ out, int use_scale) {
  __shared__ float red[256];
  int rb = blockIdx.x;
  const float* x = sim + (size_t)rb * Tdim;

  float mx = -1e30f;
  for (int j = threadIdx.x; j < Tdim; j += 256) mx = fmaxf(mx, x[j]);
  float rowmax = blk_max(mx, red);

  float se = 0.0f;
  for (int j = threadIdx.x; j < Tdim; j += 256) se += __expf(x[j] - rowmax);
  float denom = blk_sum(se, red);
  float inv_d = 1.0f / denom;

  for (int j = threadIdx.x; j < Tdim; j += 256) {
    float p = __expf(x[j] - rowmax) * inv_d;
    if (use_scale) p *= scale[j];
    out[(size_t)rb * Tdim + j] = p;
  }
}

// new_memory[b,i,j] = (1 - ww[b,j]*erase[b,i,j])*mem[i,j] + ww[b,j]*add[b,i,j]
__global__ void mem_update(const float* __restrict__ mem,
                           const float* __restrict__ WW,
                           const float* __restrict__ ER,
                           const float* __restrict__ AD,
                           float* __restrict__ out) {
  size_t idx = (size_t)blockIdx.x * 256 + threadIdx.x;
  if (idx < (size_t)Bdim * Mdim * Mdim) {
    int j = (int)(idx & (Mdim - 1));
    size_t bi = idx >> 10;
    int i = (int)(bi & (Mdim - 1));
    int b = (int)(bi >> 10);
    float w = WW[(size_t)b * Tdim + j];
    float e = ER[((size_t)b * Tdim + i) * Mdim + j];
    float a = AD[((size_t)b * Tdim + i) * Mdim + j];
    out[idx] = (1.0f - w * e) * mem[(size_t)i * Mdim + j] + w * a;
  }
}

// ---------------------------------------------------------------------------
extern "C" void kernel_launch(void* const* d_in, const int* in_sizes, int n_in,
                              void* d_out, int out_size, void* d_ws, size_t ws_size,
                              hipStream_t stream) {
  const float* inputs = (const float*)d_in[0];   // (B,T,D)
  const float* memory = (const float*)d_in[1];   // (M,M)
  const float* Wx     = (const float*)d_in[2];   // (D,4H)
  const float* Wh     = (const float*)d_in[3];   // (H,4H)
  const float* b      = (const float*)d_in[4];   // (4H,)
  const float* rk_W   = (const float*)d_in[5];   // (R,H,M)
  const float* rk_b   = (const float*)d_in[6];   // (R,M)
  const float* wk_W   = (const float*)d_in[7];   // (H,M)
  const float* wk_b   = (const float*)d_in[8];
  const float* we_W   = (const float*)d_in[9];
  const float* we_b   = (const float*)d_in[10];
  const float* wa_W   = (const float*)d_in[11];
  const float* wa_b   = (const float*)d_in[12];

  float* out_reads = (float*)d_out;                       // (R,B,T) = 32768
  float* out_mem   = (float*)d_out + (size_t)Rdim * Bdim * Tdim;

  const size_t BT = (size_t)Bdim * Tdim;                  // 8192
  float* ws = (float*)d_ws;
  size_t off = 0;
  float*     XZ    = ws + off; off += BT * 4 * Hdim;              // 16M f
  float*     H     = ws + off; off += BT * Hdim;                  // 4M f
  _Float16*  WhT   = (_Float16*)(ws + off); off += (size_t)Hdim * 4 * Hdim / 2;
  float*     RK    = ws + off; off += (size_t)Rdim * BT * Mdim;   // 32M f
  float*     WK    = ws + off; off += BT * Mdim;                  // 8M f
  float*     ER    = ws + off; off += BT * Mdim;
  float*     AD    = ws + off; off += BT * Mdim;
  float*     SIM   = ws + off; off += (size_t)Rdim * BT;          // 32768
  float*     SIMW  = ws + off; off += BT;                         // 8192
  float*     INVN  = ws + off; off += Mdim;
  float*     CSUM  = ws + off; off += Mdim;
  float*     WW    = ws + off; off += (size_t)Bdim * Tdim;

  dim3 blk(256);

  // 1. XZ = inputs @ Wx + b     (8192x256)(256x2048)
  gemm_wmma_f16<<<dim3(BT / 32, (4 * Hdim) / 128), blk, 0, stream>>>(
      inputs, Wx, b, XZ, (int)BT, 4 * Hdim, Ddim, 0);

  // 2. Wh -> f16 transposed (fragment-ready)
  cvt_transpose_f16<<<(Hdim * 4 * Hdim + 255) / 256, blk, 0, stream>>>(Wh, WhT);

  // 3. sequential LSTM (persistent single block, WMMA recurrence)
  lstm_seq<<<1, blk, 0, stream>>>(XZ, WhT, H);

  // 4. bulk GEMMs on h
  for (int r = 0; r < Rdim; ++r)
    gemm_wmma_f16<<<dim3(BT / 32, Mdim / 128), blk, 0, stream>>>(
        H, rk_W + (size_t)r * Hdim * Mdim, rk_b + (size_t)r * Mdim,
        RK + (size_t)r * BT * Mdim, (int)BT, Mdim, Hdim, 0);
  gemm_wmma_f16<<<dim3(BT / 32, Mdim / 128), blk, 0, stream>>>(
      H, wk_W, wk_b, WK, (int)BT, Mdim, Hdim, 0);
  gemm_wmma_f16<<<dim3(BT / 32, Mdim / 128), blk, 0, stream>>>(
      H, we_W, we_b, ER, (int)BT, Mdim, Hdim, 1);   // sigmoid fused
  gemm_wmma_f16<<<dim3(BT / 32, Mdim / 128), blk, 0, stream>>>(
      H, wa_W, wa_b, AD, (int)BT, Mdim, Hdim, 0);

  // 5. memory row norms + column sums
  mem_invnorm<<<Mdim, blk, 0, stream>>>(memory, INVN);
  mem_colsum<<<(Mdim + 255) / 256, blk, 0, stream>>>(memory, CSUM);

  // 6. fused softmax/l2norm/dot -> sim
  softmax_l2_dot<<<Rdim * (int)BT, blk, 0, stream>>>(RK, memory, INVN, SIM);
  softmax_l2_dot<<<(int)BT, blk, 0, stream>>>(WK, memory, INVN, SIMW);

  // 7. softmax over T: reads (scaled by colsum) and write weights ww
  softmax_T<<<Rdim * Bdim, blk, 0, stream>>>(SIM, CSUM, out_reads, 1);
  softmax_T<<<Bdim, blk, 0, stream>>>(SIMW, CSUM, WW, 0);

  // 8. memory update
  mem_update<<<(int)(((size_t)Bdim * Mdim * Mdim + 255) / 256), blk, 0, stream>>>(
      memory, WW, ER, AD, out_mem);
}
